// VectorQuantizerEMA_30202210025599
// MI455X (gfx1250) — matrix operand order repaired and checked by reading, not compiled, not run on hardware
//
#include <hip/hip_runtime.h>
#include <hip/hip_bf16.h>

#define DECAY 0.99f
#define EPSV  1e-5f
#define KCODES 1024
#define DDIM   64
#define BROWS  131072
#define MROWS  4                 // 16-row tiles per wave
#define EPAD   72                // padded LDS row stride in halfs (144 B = 36 banks)
#define SMEM_BYTES (KCODES * EPAD * 2)

typedef __attribute__((ext_vector_type(16))) _Float16 v16h;
typedef __attribute__((ext_vector_type(8)))  _Float16 v8h;
typedef __attribute__((ext_vector_type(8)))  float    v8f;
typedef __attribute__((ext_vector_type(2)))  float    v2f;
typedef int v4i __attribute__((vector_size(16)));           // matches builtin param
typedef __attribute__((address_space(1))) v4i* as1_v4i_p;   // global
typedef __attribute__((address_space(3))) v4i* as3_v4i_p;   // LDS

#if defined(__has_builtin)
#if __has_builtin(__builtin_amdgcn_global_load_async_to_lds_b128) && \
    __has_builtin(__builtin_amdgcn_s_wait_asynccnt)
#define USE_ASYNC_LDS 1
#endif
#endif

__device__ __forceinline__ void atomic_add_f32(float* p, float v) {
    __hip_atomic_fetch_add(p, v, __ATOMIC_RELAXED, __HIP_MEMORY_SCOPE_AGENT);
}

// ---------------------------------------------------------------------------
__global__ void zero_f32(float* __restrict__ p, int n) {
    int i = blockIdx.x * blockDim.x + threadIdx.x;
    if (i < n) p[i] = 0.0f;
}

// ---------------------------------------------------------------------------
// Row-normalize (x / max(||x||, eps)); wave32 per row; writes f16 copy + norm.
// ---------------------------------------------------------------------------
__global__ void __launch_bounds__(256)
normalize_rows_f16(const float* __restrict__ in,
                   _Float16* __restrict__ outh,
                   float* __restrict__ outnorm,
                   int nrows) {
    int wave = (blockIdx.x * blockDim.x + threadIdx.x) >> 5;
    int lane = threadIdx.x & 31;
    if (wave >= nrows) return;
    const float* p = in + (size_t)wave * DDIM + lane * 2;
    float x0 = p[0], x1 = p[1];
    float s = x0 * x0 + x1 * x1;
    #pragma unroll
    for (int m = 16; m >= 1; m >>= 1) s += __shfl_xor(s, m, 32);
    float zn  = fmaxf(sqrtf(s), EPSV);
    float inv = 1.0f / zn;
    _Float16* q = outh + (size_t)wave * DDIM + lane * 2;
    q[0] = (_Float16)(x0 * inv);
    q[1] = (_Float16)(x1 * inv);
    if (outnorm != nullptr && lane == 0) outnorm[wave] = zn;
}

// ---------------------------------------------------------------------------
// sim = z_n @ e_n.T via v_wmma_f32_16x16x32_f16. Codebook staged in LDS
// (padded stride to dodge bank conflicts). 4 row-tiles per wave -> 8 WMMAs
// per B-fragment load with 4 independent accumulator chains.
// ---------------------------------------------------------------------------
__global__ void __launch_bounds__(256)
sim_argmax_scatter(const _Float16* __restrict__ zh,
                   const _Float16* __restrict__ eh,
                   const float* __restrict__ ze,
                   const float* __restrict__ znorm,
                   int*   __restrict__ codes,
                   float* __restrict__ counts,
                   float* __restrict__ dw) {
    extern __shared__ _Float16 smem[];   // KCODES rows x EPAD halfs

    // ---- cooperative stage-in of the f16 codebook (1024 x 64 halfs) ----
    {
        const int NCHUNK = KCODES * 8;   // 16-byte chunks
        for (int c = threadIdx.x; c < NCHUNK; c += 256) {
            int row  = c >> 3;
            int part = c & 7;
            const _Float16* g = eh + (size_t)row * DDIM + part * 8;
            _Float16*       l = smem + row * EPAD + part * 8;
#ifdef USE_ASYNC_LDS
            __builtin_amdgcn_global_load_async_to_lds_b128(
                (as1_v4i_p)g, (as3_v4i_p)l, 0, 0);
#else
            *(v8h*)l = *(const v8h*)g;
#endif
        }
#ifdef USE_ASYNC_LDS
        __builtin_amdgcn_s_wait_asynccnt(0);
#endif
        __syncthreads();
    }

    int wave = threadIdx.x >> 5;
    int lane = threadIdx.x & 31;
    int r    = lane & 15;     // M for A, N for B/C/D
    int half = lane >> 4;
    int base = (blockIdx.x * 8 + wave) * (MROWS * 16);

    // ---- A fragments for MROWS 16-row tiles (K-interleaved per ISA layout) ----
    v16h a0[MROWS], a1[MROWS];
    #pragma unroll
    for (int m = 0; m < MROWS; ++m) {
        const _Float16* pa = zh + (size_t)(base + m * 16 + r) * DDIM;
        v8h a00 = *(const v8h*)(pa + 8 * half);
        v8h a01 = *(const v8h*)(pa + 16 + 8 * half);
        v8h a10 = *(const v8h*)(pa + 32 + 8 * half);
        v8h a11 = *(const v8h*)(pa + 48 + 8 * half);
        a0[m] = __builtin_shufflevector(a00, a01, 0,1,2,3,4,5,6,7,8,9,10,11,12,13,14,15);
        a1[m] = __builtin_shufflevector(a10, a11, 0,1,2,3,4,5,6,7,8,9,10,11,12,13,14,15);
    }

    float bestv[MROWS][8];
    int   bestc[MROWS][8];
    #pragma unroll
    for (int m = 0; m < MROWS; ++m)
        #pragma unroll
        for (int i = 0; i < 8; ++i) { bestv[m][i] = -1e30f; bestc[m][i] = 0; }

    // ---- 64 column tiles of 16 codes; D=64 -> 2 K-slices ----
    for (int t = 0; t < KCODES / 16; ++t) {
        int col = t * 16 + r;
        const _Float16* pb = smem + col * EPAD;   // B[k][n] = e_n[col][k]
        v16h b0 = __builtin_shufflevector(*(const v8h*)(pb + 8 * half),
                                          *(const v8h*)(pb + 16 + 8 * half),
                                          0,1,2,3,4,5,6,7,8,9,10,11,12,13,14,15);
        v16h b1 = __builtin_shufflevector(*(const v8h*)(pb + 32 + 8 * half),
                                          *(const v8h*)(pb + 48 + 8 * half),
                                          0,1,2,3,4,5,6,7,8,9,10,11,12,13,14,15);
        v8f c[MROWS];
        #pragma unroll
        for (int m = 0; m < MROWS; ++m)
            c[m] = __builtin_amdgcn_wmma_f32_16x16x32_f16(false, a0[m], false, b0,
                                                          (short)0, (v8f){},
                                                          false, false);
        #pragma unroll
        for (int m = 0; m < MROWS; ++m)
            c[m] = __builtin_amdgcn_wmma_f32_16x16x32_f16(false, a1[m], false, b1,
                                                          (short)0, c[m],
                                                          false, false);
        #pragma unroll
        for (int m = 0; m < MROWS; ++m)
            #pragma unroll
            for (int i = 0; i < 8; ++i)
                if (c[m][i] > bestv[m][i]) { bestv[m][i] = c[m][i]; bestc[m][i] = col; }
    }

    // ---- cross-lane argmax within each 16-lane half group ----
    #pragma unroll
    for (int m = 0; m < MROWS; ++m)
        #pragma unroll
        for (int i = 0; i < 8; ++i)
            #pragma unroll
            for (int msk = 8; msk >= 1; msk >>= 1) {
                float ov = __shfl_xor(bestv[m][i], msk, 32);
                int   oc = __shfl_xor(bestc[m][i], msk, 32);
                if (ov > bestv[m][i] ||
                    (ov == bestv[m][i] && oc < bestc[m][i])) {
                    bestv[m][i] = ov; bestc[m][i] = oc;
                }
            }

    if (r == 0) {
        #pragma unroll
        for (int m = 0; m < MROWS; ++m)
            #pragma unroll
            for (int i = 0; i < 8; ++i)
                codes[base + m * 16 + half * 8 + i] = bestc[m][i];
    }

    // ---- EMA scatter: counts[code] += 1 ; dw[code] += z_n[row] ----
    #pragma unroll
    for (int m = 0; m < MROWS; ++m) {
        #pragma unroll
        for (int rr = 0; rr < 16; ++rr) {
            int src = (rr < 8) ? 0 : 16;
            int cde = __shfl(bestc[m][rr & 7], src, 32);
            int row = base + m * 16 + rr;
            float inv = 1.0f / znorm[row];
            v2f z2 = *(const v2f*)(ze + (size_t)row * DDIM + lane * 2);
            atomic_add_f32(&dw[(size_t)cde * DDIM + lane * 2 + 0], z2.x * inv);
            atomic_add_f32(&dw[(size_t)cde * DDIM + lane * 2 + 1], z2.y * inv);
            if (lane == 0) atomic_add_f32(&counts[cde], 1.0f);
        }
    }
}

// ---------------------------------------------------------------------------
// Codebook EMA update + Laplace smoothing + normalize. K=1024, one block.
// ---------------------------------------------------------------------------
__global__ void __launch_bounds__(1024)
update_codebook(const float* __restrict__ ema_cs,
                const float* __restrict__ ema_w,
                const float* __restrict__ counts,
                const float* __restrict__ dw,
                float* __restrict__ embed_unit) {
    __shared__ float red[1024];
    int k = threadIdx.x;
    float ncs = ema_cs[k] * DECAY + counts[k] * (1.0f - DECAY);
    red[k] = ncs;
    __syncthreads();
    for (int s = 512; s > 0; s >>= 1) {
        if (k < s) red[k] += red[k + s];
        __syncthreads();
    }
    float n  = red[0];
    float cs = (ncs + EPSV) / (n + KCODES * EPSV) * n;
    float denom = fmaxf(cs, EPSV);
    float sq = 0.0f;
    for (int d = 0; d < DDIM; ++d) {
        float nw = ema_w[k * DDIM + d] * DECAY + dw[k * DDIM + d] * (1.0f - DECAY);
        float m  = nw / denom;
        sq += m * m;
    }
    float inv = 1.0f / fmaxf(sqrtf(sq), EPSV);
    for (int d = 0; d < DDIM; ++d) {
        float nw = ema_w[k * DDIM + d] * DECAY + dw[k * DDIM + d] * (1.0f - DECAY);
        embed_unit[k * DDIM + d] = (nw / denom) * inv;
    }
}

// ---------------------------------------------------------------------------
// Gather updated rows; z_q_st value == z_q; dists = 1 - <z_n, normalize(e)>.
// ---------------------------------------------------------------------------
__global__ void __launch_bounds__(256)
finalize_out(const float* __restrict__ ze,
             const float* __restrict__ znorm,
             const int*   __restrict__ codes,
             const float* __restrict__ embed_unit,
             float* __restrict__ zq_out,
             float* __restrict__ dists) {
    int row  = (blockIdx.x * blockDim.x + threadIdx.x) >> 5;
    int lane = threadIdx.x & 31;
    int c    = codes[row];
    float zn = znorm[row];
    const float* pe = embed_unit + (size_t)c * DDIM + lane * 2;
    const float* pz = ze + (size_t)row * DDIM + lane * 2;
    float e0 = pe[0], e1 = pe[1];
    float z0 = pz[0], z1 = pz[1];
    zq_out[(size_t)row * DDIM + lane * 2 + 0] = e0 * zn;
    zq_out[(size_t)row * DDIM + lane * 2 + 1] = e1 * zn;
    float s1 = z0 * e0 + z1 * e1;
    float s2 = e0 * e0 + e1 * e1;
    #pragma unroll
    for (int m = 16; m >= 1; m >>= 1) {
        s1 += __shfl_xor(s1, m, 32);
        s2 += __shfl_xor(s2, m, 32);
    }
    if (lane == 0) {
        float en = fmaxf(sqrtf(s2), EPSV);
        dists[row] = 1.0f - s1 / (zn * en);
    }
}

// ---------------------------------------------------------------------------
extern "C" void kernel_launch(void* const* d_in, const int* in_sizes, int n_in,
                              void* d_out, int out_size, void* d_ws, size_t ws_size,
                              hipStream_t stream) {
    (void)in_sizes; (void)n_in; (void)out_size; (void)ws_size;
    const float* z_e       = (const float*)d_in[0];
    const float* embedding = (const float*)d_in[1];
    const float* ema_cs    = (const float*)d_in[2];
    const float* ema_w     = (const float*)d_in[3];
    // d_in[4] = tau : argmax invariant to positive tau; unused.

    char* ws = (char*)d_ws;
    size_t off = 0;
    auto take = [&](size_t bytes) {
        void* p = ws + off;
        off += (bytes + 255) & ~(size_t)255;
        return p;
    };
    _Float16* z_h     = (_Float16*)take((size_t)BROWS * DDIM * 2);
    _Float16* e_h     = (_Float16*)take((size_t)KCODES * DDIM * 2);
    float*    znorm   = (float*)   take((size_t)BROWS * 4);
    float*    counts  = (float*)   take((size_t)KCODES * 4);        // contiguous
    float*    dw      = (float*)   take((size_t)KCODES * DDIM * 4); //  with counts
    float*    embed_u = (float*)   take((size_t)KCODES * DDIM * 4);

    float* zq    = (float*)d_out;
    int*   codes = (int*)d_out + (size_t)BROWS * DDIM;
    float* dists = (float*)d_out + (size_t)BROWS * DDIM + BROWS;

    int nzero = KCODES + KCODES * DDIM;
    zero_f32<<<(nzero + 255) / 256, 256, 0, stream>>>(counts, nzero);

    normalize_rows_f16<<<BROWS / 8, 256, 0, stream>>>(z_e, z_h, znorm, BROWS);
    normalize_rows_f16<<<KCODES / 8, 256, 0, stream>>>(embedding, e_h, nullptr, KCODES);

    // 8 waves/block, MROWS 16-row tiles per wave -> 512 rows per block
    sim_argmax_scatter<<<BROWS / (8 * MROWS * 16), 256, SMEM_BYTES, stream>>>(
        z_h, e_h, z_e, znorm, codes, counts, dw);

    update_codebook<<<1, 1024, 0, stream>>>(ema_cs, ema_w, counts, dw, embed_u);

    finalize_out<<<BROWS / 8, 256, 0, stream>>>(z_e, znorm, codes, embed_u,
                                                zq, dists);
}
